// CrossModalAttention_21543555956900
// MI455X (gfx1250) — matrix operand order repaired and checked by reading, not compiled, not run on hardware
//
#include <hip/hip_runtime.h>

typedef __bf16 bf16_t;
typedef __attribute__((ext_vector_type(16))) __bf16 v16bf;
typedef __attribute__((ext_vector_type(8)))  float  v8f;

#define HID 768
#define KVTILE_ELEMS (32 * HID)   // one 32-row bf16 tile

__device__ inline v8f wmma_bf16(v16bf a, v16bf b, v8f c) {
  return __builtin_amdgcn_wmma_f32_16x16x32_bf16(
      false, a, false, b, (short)0, c, false, false);
}

// ---- CDNA5 async global->LDS (ASYNCcnt-tracked DMA, no VGPR staging) ----
__device__ inline void async_copy_b128(unsigned lds_off, const void* gptr) {
  asm volatile("global_load_async_to_lds_b128 %0, %1, off"
               :: "v"(lds_off), "v"((unsigned long long)(uintptr_t)gptr)
               : "memory");
}
__device__ inline void wait_async0() {
#if __has_builtin(__builtin_amdgcn_s_wait_asynccnt)
  __builtin_amdgcn_s_wait_asynccnt(0);
#else
  asm volatile("s_wait_asynccnt 0x0" ::: "memory");
#endif
}
// Copy a 32x768 bf16 tile (49152 B = 3072 x b128) global -> LDS, 12 per thread.
__device__ inline void async_tile_load(bf16_t* lds, const bf16_t* g, int tid) {
  unsigned l0 = (unsigned)(uintptr_t)lds;   // low 32 bits of shared aperture = LDS offset
  const char* gp = (const char*)g;
#pragma unroll
  for (int i = 0; i < 12; ++i) {
    int byteoff = (tid + i * 256) * 16;
    async_copy_b128(l0 + byteoff, gp + byteoff);
  }
}

// A fragment (16x32 bf16) from LDS, row-major buf[row][k].
__device__ inline v16bf load_a_frag(const bf16_t* buf, int row0, int k0,
                                    int stride, int lane) {
  v16bf a;
  int m  = lane & 15;
  int kb = (lane >> 4) << 4; // 0 or 16
  const bf16_t* p = buf + (row0 + m) * stride + k0;
#pragma unroll
  for (int v = 0; v < 4; ++v) {
    a[2 * v]     = p[kb + 2 * v];
    a[2 * v + 1] = p[kb + 2 * v + 1];
  }
#pragma unroll
  for (int v = 0; v < 4; ++v) {
    a[8 + 2 * v]     = p[8 + kb + 2 * v];
    a[8 + 2 * v + 1] = p[8 + kb + 2 * v + 1];
  }
  return a;
}

// B fragment (32x16) from B^T layout buf[n][k] (k contiguous).
__device__ inline v16bf load_bT_frag(const bf16_t* buf, int col0, int k0,
                                     int stride, int lane) {
  v16bf b;
  int n  = lane & 15;
  int kb = (lane >> 4) << 4; // 0 or 16
  const bf16_t* p = buf + (col0 + n) * stride + k0 + kb;
#pragma unroll
  for (int v = 0; v < 8; ++v) {
    b[2 * v]     = p[2 * v];
    b[2 * v + 1] = p[2 * v + 1];
  }
  return b;
}

// B fragment (32x16) from natural layout buf[k][n].
__device__ inline v16bf load_b_frag(const bf16_t* buf, int k0, int col0,
                                    int stride, int lane) {
  v16bf b;
  int n  = lane & 15;
  int kb = (lane >> 4) << 4; // 0 or 16
#pragma unroll
  for (int v = 0; v < 8; ++v) {
    b[2 * v]     = buf[(k0 + kb + 2 * v) * stride + col0 + n];
    b[2 * v + 1] = buf[(k0 + kb + 2 * v + 1) * stride + col0 + n];
  }
  return b;
}

// ---------------------------------------------------------------------------
// Projection GEMM: out[m][n] = bf16( sum_k X[m][k]*W[n][k] + bias[n] )
// ---------------------------------------------------------------------------
__global__ __launch_bounds__(256) void proj_gemm(
    const float* __restrict__ X, const float* __restrict__ W,
    const float* __restrict__ bias, bf16_t* __restrict__ out,
    int N, int Kdim) {
  __shared__ bf16_t As[64 * 32];
  __shared__ bf16_t Bs[64 * 32];

  const int tid  = threadIdx.x;
  const int lane = tid & 31;
  const int wave = tid >> 5;
  const int m0   = blockIdx.x * 64;
  const int n0   = blockIdx.y * 64;
  const int mt   = wave & 3;
  const int ntb  = (wave >> 2) * 2;

  v8f acc0 = {}, acc1 = {};

  for (int k0 = 0; k0 < Kdim; k0 += 32) {
#pragma unroll
    for (int i = 0; i < 2; ++i) {
      int idx = tid + i * 256;
      int r = idx >> 3;
      int c = (idx & 7) * 4;
      float4 f = *(const float4*)(X + (size_t)(m0 + r) * Kdim + k0 + c);
      bf16_t* d = As + r * 32 + c;
      d[0] = (bf16_t)f.x; d[1] = (bf16_t)f.y;
      d[2] = (bf16_t)f.z; d[3] = (bf16_t)f.w;
    }
#pragma unroll
    for (int i = 0; i < 2; ++i) {
      int idx = tid + i * 256;
      int r = idx >> 3;
      int c = (idx & 7) * 4;
      float4 f = *(const float4*)(W + (size_t)(n0 + r) * Kdim + k0 + c);
      bf16_t* d = Bs + r * 32 + c;
      d[0] = (bf16_t)f.x; d[1] = (bf16_t)f.y;
      d[2] = (bf16_t)f.z; d[3] = (bf16_t)f.w;
    }
    __syncthreads();

    v16bf a  = load_a_frag(As, mt * 16, 0, 32, lane);
    v16bf b0 = load_bT_frag(Bs, ntb * 16, 0, 32, lane);
    v16bf b1 = load_bT_frag(Bs, ntb * 16 + 16, 0, 32, lane);
    acc0 = wmma_bf16(a, b0, acc0);
    acc1 = wmma_bf16(a, b1, acc1);
    __syncthreads();
  }

  const int mrow  = mt * 16 + ((lane >> 4) << 3);
  const int ncol0 = n0 + ntb * 16 + (lane & 15);
#pragma unroll
  for (int r = 0; r < 8; ++r) {
    int gm = m0 + mrow + r;
    out[(size_t)gm * N + ncol0]      = (bf16_t)(acc0[r] + bias[ncol0]);
    out[(size_t)gm * N + ncol0 + 16] = (bf16_t)(acc1[r] + bias[ncol0 + 16]);
  }
}

// ---------------------------------------------------------------------------
// Flash attention (unscaled logits), 768-dim single head, bf16 WMMA.
// 256 threads = 8 waves; q-tile 32 rows; kv-tile 32 keys, double-buffered
// K/V tiles fed by async global->LDS DMA (ASYNCcnt).
// ---------------------------------------------------------------------------
__global__ __launch_bounds__(256) void attn_kernel(
    const bf16_t* __restrict__ Q, const bf16_t* __restrict__ K,
    const bf16_t* __restrict__ V, float* __restrict__ out,
    int qlen, int kvlen) {
  extern __shared__ bf16_t smem[];
  bf16_t* Qs   = smem;                      // 32*768
  bf16_t* Kbuf = Qs + KVTILE_ELEMS;         // 2 x 32*768
  bf16_t* Vbuf = Kbuf + 2 * KVTILE_ELEMS;   // 2 x 32*768
  float*  sc0  = (float*)(Vbuf + 2 * KVTILE_ELEMS); // 32*32
  float*  sc1  = sc0 + 32 * 32;                     // 32*32
  bf16_t* Ps   = (bf16_t*)(sc1 + 32 * 32);          // 32*32
  float*  mrow = (float*)(Ps + 32 * 32);            // 32
  float*  lrow = mrow + 32;                         // 32
  float*  srow = lrow + 32;                         // 32

  const int tid  = threadIdx.x;
  const int lane = tid & 31;
  const int wave = tid >> 5;
  const int nqt  = qlen / 32;
  const int b    = blockIdx.x / nqt;
  const int qt   = blockIdx.x % nqt;

  const bf16_t* Qb = Q + ((size_t)b * qlen + qt * 32) * HID;
  const bf16_t* Kb = K + (size_t)b * kvlen * HID;
  const bf16_t* Vb = V + (size_t)b * kvlen * HID;

  // Kick off DMA: Q tile + kv tile 0 (buffer 0)
  async_tile_load(Qs, Qb, tid);
  async_tile_load(Kbuf, Kb, tid);
  async_tile_load(Vbuf, Vb, tid);

  if (tid < 32) { mrow[tid] = -1e30f; lrow[tid] = 0.0f; srow[tid] = 1.0f; }

  v8f o[2][6];
#pragma unroll
  for (int i = 0; i < 2; ++i)
#pragma unroll
    for (int j = 0; j < 6; ++j) o[i][j] = (v8f){};

  const int smt   = (wave >> 1) & 1;
  const int snt   = wave & 1;
  const int khalf = wave >> 2;
  float* scp = khalf ? sc1 : sc0;

  const int niter = kvlen / 32;
  for (int it = 0; it < niter; ++it) {
    bf16_t* Kc = Kbuf + (it & 1) * KVTILE_ELEMS;
    bf16_t* Vc = Vbuf + (it & 1) * KVTILE_ELEMS;

    wait_async0();     // my DMA for this tile (and Q on it==0) has landed
    __syncthreads();   // everyone's has landed; prev iter reads finished

    // Prefetch next kv tile into the other buffer; DMA overlaps compute.
    if (it + 1 < niter) {
      bf16_t* Kn = Kbuf + ((it + 1) & 1) * KVTILE_ELEMS;
      bf16_t* Vn = Vbuf + ((it + 1) & 1) * KVTILE_ELEMS;
      async_tile_load(Kn, Kb + (size_t)(it + 1) * 32 * HID, tid);
      async_tile_load(Vn, Vb + (size_t)(it + 1) * 32 * HID, tid);
    }

    // ---- S = Q . K^T partial over this wave's 384-deep k-half ----
    v8f s = {};
#pragma unroll
    for (int kk = 0; kk < 12; ++kk) {
      int k0 = khalf * 384 + kk * 32;
      v16bf a  = load_a_frag(Qs, smt * 16, k0, HID, lane);
      v16bf bb = load_bT_frag(Kc, snt * 16, k0, HID, lane);
      s = wmma_bf16(a, bb, s);
    }
    {
      int m = smt * 16 + ((lane >> 4) << 3);
      int n = snt * 16 + (lane & 15);
#pragma unroll
      for (int r = 0; r < 8; ++r) scp[(m + r) * 32 + n] = s[r];
    }
    __syncthreads();

    // ---- Online softmax: thread t owns q-row t ----
    if (tid < 32) {
      float mo = mrow[tid];
      float mx = mo;
      float sv[32];
#pragma unroll
      for (int j = 0; j < 32; ++j) {
        sv[j] = sc0[tid * 32 + j] + sc1[tid * 32 + j];
        mx = fmaxf(mx, sv[j]);
      }
      float scl = __expf(mo - mx);
      float ls  = 0.0f;
#pragma unroll
      for (int j = 0; j < 32; ++j) {
        float p = __expf(sv[j] - mx);
        ls += p;
        Ps[tid * 32 + j] = (bf16_t)p;
      }
      mrow[tid] = mx;
      lrow[tid] = lrow[tid] * scl + ls;
      srow[tid] = scl;
    }
    __syncthreads();

    // ---- Rescale O, accumulate O += P . V ----
    v16bf pa0 = load_a_frag(Ps, 0, 0, 32, lane);
    v16bf pa1 = load_a_frag(Ps, 16, 0, 32, lane);
#pragma unroll
    for (int mt2 = 0; mt2 < 2; ++mt2) {
      int mb = mt2 * 16 + ((lane >> 4) << 3);
#pragma unroll
      for (int j = 0; j < 6; ++j)
#pragma unroll
        for (int r = 0; r < 8; ++r) o[mt2][j][r] *= srow[mb + r];
    }
#pragma unroll
    for (int j = 0; j < 6; ++j) {
      int col = wave * 96 + j * 16;
      v16bf vb = load_b_frag(Vc, 0, col, HID, lane);
      o[0][j] = wmma_bf16(pa0, vb, o[0][j]);
      o[1][j] = wmma_bf16(pa1, vb, o[1][j]);
    }
    __syncthreads();
  }

  // ---- Normalize by l, store f32 ----
  float* ob = out + ((size_t)b * qlen + qt * 32) * HID;
#pragma unroll
  for (int mt2 = 0; mt2 < 2; ++mt2) {
    int mb = mt2 * 16 + ((lane >> 4) << 3);
#pragma unroll
    for (int j = 0; j < 6; ++j) {
      int col = wave * 96 + j * 16 + (lane & 15);
#pragma unroll
      for (int r = 0; r < 8; ++r) {
        ob[(size_t)(mb + r) * HID + col] = o[mt2][j][r] / lrow[mb + r];
      }
    }
  }
}

extern "C" void kernel_launch(void* const* d_in, const int* in_sizes, int n_in,
                              void* d_out, int out_size, void* d_ws, size_t ws_size,
                              hipStream_t stream) {
  (void)in_sizes; (void)n_in; (void)out_size; (void)ws_size;
  const float* x  = (const float*)d_in[0];
  const float* kv = (const float*)d_in[1];
  const float* Wq = (const float*)d_in[2];
  const float* bq = (const float*)d_in[3];
  const float* Wk = (const float*)d_in[4];
  const float* bk = (const float*)d_in[5];
  const float* Wv = (const float*)d_in[6];
  const float* bv = (const float*)d_in[7];
  float* out = (float*)d_out;

  const int B = 16, QL = 2048, KL = 2048, QD = 768, KD = 1024, H = HID;
  const int Mq = B * QL, Mk = B * KL;

  bf16_t* Qw = (bf16_t*)d_ws;
  bf16_t* Kw = Qw + (size_t)Mq * H;
  bf16_t* Vw = Kw + (size_t)Mk * H;

  dim3 blk(256);
  proj_gemm<<<dim3(Mq / 64, H / 64), blk, 0, stream>>>(x,  Wq, bq, Qw, H, QD);
  proj_gemm<<<dim3(Mk / 64, H / 64), blk, 0, stream>>>(kv, Wk, bk, Kw, H, KD);
  proj_gemm<<<dim3(Mk / 64, H / 64), blk, 0, stream>>>(kv, Wv, bv, Vw, H, KD);

  size_t shmem = (size_t)(5 * KVTILE_ELEMS) * sizeof(bf16_t)  // Q + 2xK + 2xV
               + (size_t)(2 * 32 * 32) * sizeof(float)        // S partials
               + (size_t)(32 * 32) * sizeof(bf16_t)           // P
               + (size_t)(3 * 32) * sizeof(float);            // m, l, scale
  hipFuncSetAttribute((const void*)attn_kernel,
                      hipFuncAttributeMaxDynamicSharedMemorySize, (int)shmem);
  attn_kernel<<<dim3(B * (QL / 32)), blk, shmem, stream>>>(Qw, Kw, Vw, out, QL, KL);
}
